// AllostericMessagePassingEncoder_87333864997030
// MI455X (gfx1250) — compile-verified, matrix operand order checked by reading, not run on hardware
//
#include <hip/hip_runtime.h>
#include <hip/hip_bf16.h>
#include <math.h>

#define NRES 512
#define HD   128
#define NL   4
#define CUT  18.0f

typedef __attribute__((ext_vector_type(16))) __bf16 v16bf;
typedef __attribute__((ext_vector_type(8)))  float  v8f;

__device__ __forceinline__ float sigmoidf_(float x){ return 1.0f/(1.0f+__expf(-x)); }
__device__ __forceinline__ float siluf_(float x){ return x*sigmoidf_(x); }
__device__ __forceinline__ unsigned f2bf_(float f){
  unsigned u = __float_as_uint(f);
  u += 0x7fffu + ((u>>16)&1u);
  return u>>16;
}
__device__ __forceinline__ unsigned pack2bf_(float lo, float hi){
  return f2bf_(lo) | (f2bf_(hi)<<16);
}

// A-fragment (16x32 bf16 slice, k-step t) from a packed-bf16 LDS tile [16 rows][64 dwords].
// ISA 16-bit A layout: lanes 0-15 m=lane, K 0..7 (v0-3) / 16..23 (v4-7);
// lanes 16-31 same m, K 8..15 / 24..31.
__device__ __forceinline__ v16bf load_a_frag(const unsigned* sbf, int t, int lane){
  const int m = lane & 15, half = lane >> 4;
  const unsigned* row = sbf + m*64 + t*16 + half*4;
  uint4 a = *(const uint4*)row;
  uint4 b = *(const uint4*)(row + 8);
  union { unsigned u[8]; v16bf v; } U;
  U.u[0]=a.x; U.u[1]=a.y; U.u[2]=a.z; U.u[3]=a.w;
  U.u[4]=b.x; U.u[5]=b.y; U.u[6]=b.z; U.u[7]=b.w;
  return U.v;
}

__device__ __forceinline__ v16bf load_b_frag(const unsigned* frag, int w, int t, int lane){
  const uint4* p = (const uint4*)(frag + (((size_t)(w*4+t))*32 + lane)*8);
  uint4 a = p[0], b = p[1];
  union { unsigned u[8]; v16bf v; } U;
  U.u[0]=a.x; U.u[1]=a.y; U.u[2]=a.z; U.u[3]=a.w;
  U.u[4]=b.x; U.u[5]=b.y; U.u[6]=b.z; U.u[7]=b.w;
  return U.v;
}

// ---------------- setup: weight swizzle to WMMA B-fragment layout + zero vec ----------------
// frag index bits: l[15:14] mat[13] w[12:10] t[9:8] lane[7:3] v[2:0]
__global__ void k_swizzle(const float* __restrict__ ew2, const float* __restrict__ vg1,
                          unsigned* __restrict__ frag, float* __restrict__ vecb){
  int id = blockIdx.x*blockDim.x + threadIdx.x;           // 65536 threads
  if (id < NRES*3) vecb[id] = 0.0f;
  int v = id & 7, lane = (id>>3)&31, t=(id>>8)&3, w=(id>>10)&7, mat=(id>>13)&1, l=(id>>14)&3;
  int nn = lane & 15, half = lane >> 4;
  int n  = w*16 + nn;
  int k0 = (v < 4) ? (t*32 + half*8 + 2*v) : (t*32 + 16 + half*8 + 2*(v-4));
  const float* src = (mat ? vg1 : ew2) + ((size_t)l*HD + k0)*HD + n;
  frag[id] = pack2bf_(src[0], src[HD]);
}

// ---------------- node init ----------------
__global__ void k_node0(const float* __restrict__ emb, const int* __restrict__ rt,
                        const float* __restrict__ cons, const float* __restrict__ cons_w,
                        const float* __restrict__ cons_b, const float* __restrict__ fs,
                        const float* __restrict__ state_w, const float* __restrict__ state_b,
                        float* __restrict__ node){
  int i = blockIdx.x, h = threadIdx.x;
  float v = emb[rt[i]*HD + h] + cons[i]*cons_w[h] + cons_b[h] + state_b[h];
  #pragma unroll
  for (int k=0;k<3;k++) v += fs[k]*state_w[k*HD+h];
  node[i*HD+h] = v;
}

// ---------------- per-layer Ci / Cj precompute (W1 decomposition) ----------------
__global__ void k_cicj(const float* __restrict__ node, const float* __restrict__ w1,
                       const float* __restrict__ b1, const float* __restrict__ cons,
                       const unsigned char* __restrict__ memm, const unsigned char* __restrict__ catm,
                       float* __restrict__ Ci, float* __restrict__ Cj){
  __shared__ float nrow[HD];
  int i = blockIdx.x, h = threadIdx.x;
  nrow[h] = node[i*HD+h];
  __syncthreads();
  float a = b1[h] + cons[i]*w1[257*HD+h] + (memm[i]?1.f:0.f)*w1[259*HD+h];
  float b =        cons[i]*w1[258*HD+h] + (catm[i]?1.f:0.f)*w1[260*HD+h];
  for (int k=0;k<HD;k++){
    float nv = nrow[k];
    a += nv * w1[k*HD + h];
    b += nv * w1[(HD+k)*HD + h];
  }
  Ci[i*HD+h] = a; Cj[i*HD+h] = b;
}

// ---------------- fused edge kernel (one block per i, 8 waves) ----------------
__global__ void __launch_bounds__(256)
k_edge(const float* __restrict__ pos, const float* __restrict__ Ci, const float* __restrict__ Cj,
       const float* __restrict__ w1, const float* __restrict__ b2row,
       const float* __restrict__ vgb1, const float* __restrict__ vgw2, const float* __restrict__ vgb2,
       const unsigned* __restrict__ w2frag, const unsigned* __restrict__ vg1frag,
       float* __restrict__ msg, float* __restrict__ vec, float* __restrict__ out_ew, int write_ew)
{
  __shared__ float ci_s[HD];
  __shared__ __align__(16) unsigned s_bf[16*64];   // packed bf16 16x128 tile (A-matrix source)
  __shared__ float eh_s[16*HD];
  __shared__ float ew_s[16], mask_s[16], dir_s[3*16], vg_s[16];
  __shared__ float part_s[8*16];
  __shared__ float msg_acc[HD];
  __shared__ float vec_acc[3];

  const int tid  = threadIdx.x;
  const int i    = blockIdx.x;
  const int lane = tid & 31;
  const int w    = tid >> 5;       // wave id -> output column tile
  const int nn   = lane & 15;
  const int half = lane >> 4;
  const int n    = (w<<4) + nn;    // output column owned by this lane

  // Register-resident B fragments (edge_w2, vg_w1), 4 k-steps each.
  v16bf wb0 = load_b_frag(w2frag, w, 0, lane);
  v16bf wb1 = load_b_frag(w2frag, w, 1, lane);
  v16bf wb2 = load_b_frag(w2frag, w, 2, lane);
  v16bf wb3 = load_b_frag(w2frag, w, 3, lane);
  v16bf vb0 = load_b_frag(vg1frag, w, 0, lane);
  v16bf vb1 = load_b_frag(vg1frag, w, 1, lane);
  v16bf vb2 = load_b_frag(vg1frag, w, 2, lane);
  v16bf vb3 = load_b_frag(vg1frag, w, 3, lane);

  const float b2v   = b2row[n];
  const float vgb1v = vgb1[n];
  const float vgw2v = vgw2[n];
  const float vgb2v = vgb2[0];

  if (tid < HD){ ci_s[tid] = Ci[i*HD+tid]; msg_acc[tid] = 0.f; }
  if (tid < 3) vec_acc[tid] = 0.f;

  const int r = tid >> 4, c = tid & 15, h0 = c*8;
  float wd8[8], wk8[8];
  #pragma unroll
  for (int e=0;e<8;e++){ wd8[e] = w1[256*HD + h0 + e]; wk8[e] = w1[261*HD + h0 + e]; }
  const float pix = pos[i*3+0], piy = pos[i*3+1], piz = pos[i*3+2];
  __syncthreads();

  for (int jt = 0; jt < NRES/16; ++jt){
    const int j0 = jt*16;
    const int j  = j0 + r;
    // ---- phase A: pre = Ci + Cj + d*wd + mask*wmk ; silu ; pack bf16 tile ----
    float dx = pix - pos[j*3+0], dy = piy - pos[j*3+1], dz = piz - pos[j*3+2];
    float d  = sqrtf(dx*dx + dy*dy + dz*dz);
    d = fmaxf(d, 1e-8f);
    float mk = (d <= CUT && j != i) ? 1.f : 0.f;
    if (c == 0){
      float inv = 1.f/d;
      dir_s[r] = dx*inv; dir_s[16+r] = dy*inv; dir_s[32+r] = dz*inv;
      mask_s[r] = mk;
    }
    const float* cjr = Cj + (size_t)j*HD + h0;
    if (jt + 1 < NRES/16)
      __builtin_prefetch((const void*)(Cj + (size_t)(j+16)*HD + h0), 0, 1);
    {
      unsigned pk[4];
      #pragma unroll
      for (int p=0;p<4;p++){
        float x0 = ci_s[h0+2*p]   + cjr[2*p]   + d*wd8[2*p]   + mk*wk8[2*p];
        float x1 = ci_s[h0+2*p+1] + cjr[2*p+1] + d*wd8[2*p+1] + mk*wk8[2*p+1];
        pk[p] = pack2bf_(siluf_(x0), siluf_(x1));
      }
      *(uint4*)(s_bf + r*64 + h0/2) = make_uint4(pk[0],pk[1],pk[2],pk[3]);
    }
    __syncthreads();

    // ---- phase B: GEMM1  eh = silu(pre) @ W2 + b2 ----
    {
      v16bf a0 = load_a_frag(s_bf, 0, lane);
      v16bf a1 = load_a_frag(s_bf, 1, lane);
      v16bf a2 = load_a_frag(s_bf, 2, lane);
      v16bf a3 = load_a_frag(s_bf, 3, lane);
      v8f acc = {0.f,0.f,0.f,0.f,0.f,0.f,0.f,0.f};
      acc = __builtin_amdgcn_wmma_f32_16x16x32_bf16(false, a0, false, wb0, (short)0, acc, false, false);
      acc = __builtin_amdgcn_wmma_f32_16x16x32_bf16(false, a1, false, wb1, (short)0, acc, false, false);
      acc = __builtin_amdgcn_wmma_f32_16x16x32_bf16(false, a2, false, wb2, (short)0, acc, false, false);
      acc = __builtin_amdgcn_wmma_f32_16x16x32_bf16(false, a3, false, wb3, (short)0, acc, false, false);
      float p[8];
      #pragma unroll
      for (int rr=0; rr<8; rr++){
        p[rr] = acc[rr] + b2v;
        eh_s[(half*8+rr)*HD + n] = p[rr];
      }
      // row-sum reduction for ew: Σ over this wave's 16 columns, per row.
      #pragma unroll
      for (int off=1; off<16; off<<=1){
        #pragma unroll
        for (int rr=0; rr<8; rr++) p[rr] += __shfl_xor(p[rr], off, 32);
      }
      if (nn == 0){
        #pragma unroll
        for (int rr=0; rr<8; rr++) part_s[w*16 + half*8 + rr] = p[rr];
      }
    }
    __syncthreads();

    // ---- phase C: ew = sigmoid(mean(eh))*mask ; msg += eh*ew ; eh -> bf16 tile ----
    if (tid < 16){
      float s = 0.f;
      #pragma unroll
      for (int ww=0; ww<8; ww++) s += part_s[ww*16 + tid];
      float ew = sigmoidf_(s * (1.0f/HD)) * mask_s[tid];
      ew_s[tid] = ew;
      if (write_ew) out_ew[(size_t)i*NRES + j0 + tid] = ew;
    }
    __syncthreads();
    if (tid < HD){
      float m = 0.f;
      #pragma unroll 4
      for (int rr=0; rr<16; rr++) m += eh_s[rr*HD + tid]*ew_s[rr];
      msg_acc[tid] += m;
    }
    {
      const float* er = eh_s + r*HD + h0;
      unsigned pk[4];
      #pragma unroll
      for (int p=0;p<4;p++) pk[p] = pack2bf_(er[2*p], er[2*p+1]);
      *(uint4*)(s_bf + r*64 + h0/2) = make_uint4(pk[0],pk[1],pk[2],pk[3]);
    }
    __syncthreads();

    // ---- phase D: GEMM2  g = silu(eh @ Vg1 + vb1); vgate = (g . vg_w2 + vb2)*ew ----
    {
      v16bf a0 = load_a_frag(s_bf, 0, lane);
      v16bf a1 = load_a_frag(s_bf, 1, lane);
      v16bf a2 = load_a_frag(s_bf, 2, lane);
      v16bf a3 = load_a_frag(s_bf, 3, lane);
      v8f acc2 = {0.f,0.f,0.f,0.f,0.f,0.f,0.f,0.f};
      acc2 = __builtin_amdgcn_wmma_f32_16x16x32_bf16(false, a0, false, vb0, (short)0, acc2, false, false);
      acc2 = __builtin_amdgcn_wmma_f32_16x16x32_bf16(false, a1, false, vb1, (short)0, acc2, false, false);
      acc2 = __builtin_amdgcn_wmma_f32_16x16x32_bf16(false, a2, false, vb2, (short)0, acc2, false, false);
      acc2 = __builtin_amdgcn_wmma_f32_16x16x32_bf16(false, a3, false, vb3, (short)0, acc2, false, false);
      float rp[8];
      #pragma unroll
      for (int rr=0; rr<8; rr++) rp[rr] = siluf_(acc2[rr] + vgb1v) * vgw2v;
      #pragma unroll
      for (int off=1; off<16; off<<=1){
        #pragma unroll
        for (int rr=0; rr<8; rr++) rp[rr] += __shfl_xor(rp[rr], off, 32);
      }
      if (nn == 0){
        #pragma unroll
        for (int rr=0; rr<8; rr++) part_s[w*16 + half*8 + rr] = rp[rr];
      }
    }
    __syncthreads();
    if (tid < 16){
      float tot = vgb2v;
      #pragma unroll
      for (int ww=0; ww<8; ww++) tot += part_s[ww*16 + tid];
      vg_s[tid] = tot * ew_s[tid];
    }
    __syncthreads();
    if (tid < 3){
      float a = 0.f;
      #pragma unroll
      for (int rr=0; rr<16; rr++) a += vg_s[rr]*dir_s[tid*16 + rr];
      vec_acc[tid] += a;
    }
    __syncthreads();
  }

  if (tid < HD) msg[(size_t)i*HD + tid] = msg_acc[tid];
  if (tid < 3)  vec[i*3 + tid] += vec_acc[tid];
}

// ---------------- per-layer node update + LayerNorm ----------------
__global__ void k_su(const float* __restrict__ msg, const float* __restrict__ w1,
                     const float* __restrict__ b1, const float* __restrict__ w2,
                     const float* __restrict__ b2, const float* __restrict__ g,
                     const float* __restrict__ bb, float* __restrict__ node){
  __shared__ float cat[2*HD];
  __shared__ float hid[HD];
  __shared__ float red[HD];
  int i = blockIdx.x, h = threadIdx.x;
  cat[h]    = node[(size_t)i*HD + h];
  cat[HD+h] = msg[(size_t)i*HD + h];
  __syncthreads();
  float a = b1[h];
  for (int k=0;k<2*HD;k++) a += cat[k]*w1[k*HD + h];
  hid[h] = siluf_(a);
  __syncthreads();
  float u = b2[h];
  for (int k=0;k<HD;k++) u += hid[k]*w2[k*HD + h];
  float x = cat[h] + u;
  red[h] = x; __syncthreads();
  for (int s=64;s>0;s>>=1){ if (h<s) red[h]+=red[h+s]; __syncthreads(); }
  float mu = red[0]*(1.0f/HD); __syncthreads();
  float xd = x - mu;
  red[h] = xd*xd; __syncthreads();
  for (int s=64;s>0;s>>=1){ if (h<s) red[h]+=red[h+s]; __syncthreads(); }
  float var = red[0]*(1.0f/HD);
  node[(size_t)i*HD + h] = xd * rsqrtf(var + 1e-5f) * g[h] + bb[h];
}

// ---------------- global head ----------------
__global__ void k_global(const float* __restrict__ node, const unsigned char* __restrict__ pm,
                         const unsigned char* __restrict__ mm,
                         const float* __restrict__ w1, const float* __restrict__ b1,
                         const float* __restrict__ w2, const float* __restrict__ b2,
                         float* __restrict__ out, float* __restrict__ gemb){
  __shared__ float gin[3*HD];
  __shared__ float hid[HD];
  int h = threadIdx.x;
  float s=0.f, sp=0.f, sm=0.f, cp=0.f, cm=0.f;
  for (int i2=0;i2<NRES;i2++){
    float v = node[(size_t)i2*HD + h];
    float fp = pm[i2]?1.f:0.f, fm = mm[i2]?1.f:0.f;
    s += v; sp += v*fp; sm += v*fm; cp += fp; cm += fm;
  }
  float mean = s*(1.0f/NRES);
  float pe = (cp > 0.f) ? sp/fmaxf(cp,1.f) : mean;
  float me = (cm > 0.f) ? sm/fmaxf(cm,1.f) : 0.f;
  gin[h]=mean; gin[HD+h]=pe; gin[2*HD+h]=me;
  out[329344 + h] = pe;
  out[329472 + h] = me;
  __syncthreads();
  float a = b1[h];
  for (int k=0;k<3*HD;k++) a += gin[k]*w1[k*HD + h];
  hid[h] = siluf_(a);
  __syncthreads();
  float ge = b2[h];
  for (int k=0;k<HD;k++) ge += hid[k]*w2[k*HD + h];
  out[329216 + h] = ge;
  gemb[h] = ge;
}

// ---------------- catalytic head + output copies ----------------
__global__ void k_cat(const float* __restrict__ node, const float* __restrict__ vec,
                      const float* __restrict__ gemb,
                      const unsigned char* __restrict__ pm, const unsigned char* __restrict__ mm,
                      const unsigned char* __restrict__ cm,
                      const float* __restrict__ w1, const float* __restrict__ b1,
                      const float* __restrict__ w2, const float* __restrict__ b2,
                      float* __restrict__ out){
  __shared__ float nr[HD], ge[HD], red[HD];
  int i = blockIdx.x, h = threadIdx.x;
  float nv = node[(size_t)i*HD + h];
  nr[h] = nv; ge[h] = gemb[h];
  __syncthreads();
  float a = b1[h];
  for (int k=0;k<HD;k++) a += nr[k]*w1[k*HD + h];
  for (int k=0;k<HD;k++) a += ge[k]*w1[(HD+k)*HD + h];
  a += (pm[i]?1.f:0.f)*w1[256*HD + h] + (mm[i]?1.f:0.f)*w1[257*HD + h] + (cm[i]?1.f:0.f)*w1[258*HD + h];
  red[h] = siluf_(a) * w2[h];
  __syncthreads();
  for (int s=64;s>0;s>>=1){ if (h<s) red[h]+=red[h+s]; __syncthreads(); }
  if (h == 0) out[329600 + i] = sigmoidf_(red[0] + b2[0]);
  out[(size_t)i*HD + h] = nv;
  if (h < 3) out[65536 + i*3 + h] = vec[i*3 + h];
}

extern "C" void kernel_launch(void* const* d_in, const int* in_sizes, int n_in,
                              void* d_out, int out_size, void* d_ws, size_t ws_size,
                              hipStream_t stream){
  const float* pos     = (const float*)d_in[0];
  const float* cons    = (const float*)d_in[1];
  const float* fs      = (const float*)d_in[2];
  const float* emb     = (const float*)d_in[3];
  const float* cons_w  = (const float*)d_in[4];
  const float* cons_b  = (const float*)d_in[5];
  const float* state_w = (const float*)d_in[6];
  const float* state_b = (const float*)d_in[7];
  const float* edge_w1 = (const float*)d_in[8];
  const float* edge_b1 = (const float*)d_in[9];
  const float* edge_w2 = (const float*)d_in[10];
  const float* edge_b2 = (const float*)d_in[11];
  const float* su_w1   = (const float*)d_in[12];
  const float* su_b1   = (const float*)d_in[13];
  const float* su_w2   = (const float*)d_in[14];
  const float* su_b2   = (const float*)d_in[15];
  const float* vg_w1   = (const float*)d_in[16];
  const float* vg_b1   = (const float*)d_in[17];
  const float* vg_w2   = (const float*)d_in[18];
  const float* vg_b2   = (const float*)d_in[19];
  const float* ln_g    = (const float*)d_in[20];
  const float* ln_b    = (const float*)d_in[21];
  const float* gh_w1   = (const float*)d_in[22];
  const float* gh_b1   = (const float*)d_in[23];
  const float* gh_w2   = (const float*)d_in[24];
  const float* gh_b2   = (const float*)d_in[25];
  const float* ch_w1   = (const float*)d_in[26];
  const float* ch_b1   = (const float*)d_in[27];
  const float* ch_w2   = (const float*)d_in[28];
  const float* ch_b2   = (const float*)d_in[29];
  const int*   rt      = (const int*)d_in[30];
  const unsigned char* pmk = (const unsigned char*)d_in[31];
  const unsigned char* mmk = (const unsigned char*)d_in[32];
  const unsigned char* cmk = (const unsigned char*)d_in[33];
  float* out = (float*)d_out;

  float* ws   = (float*)d_ws;
  float* node = ws;                  // 65536 f32
  float* Ci   = ws + 65536;          // 65536
  float* Cj   = ws + 131072;         // 65536
  float* msg  = ws + 196608;         // 65536
  float* vecb = ws + 262144;         // 1536
  float* gemb = ws + 263680;         // 128
  unsigned* frag = (unsigned*)(ws + 263936); // 65536 dwords (bf16 B-fragments)

  k_swizzle<<<256, 256, 0, stream>>>(edge_w2, vg_w1, frag, vecb);
  k_node0  <<<NRES, HD, 0, stream>>>(emb, rt, cons, cons_w, cons_b, fs, state_w, state_b, node);

  for (int l = 0; l < NL; ++l){
    const float* w1l = edge_w1 + (size_t)l*262*HD;
    k_cicj<<<NRES, HD, 0, stream>>>(node, w1l, edge_b1 + l*HD, cons, mmk, cmk, Ci, Cj);
    k_edge<<<NRES, 256, 0, stream>>>(pos, Ci, Cj, w1l, edge_b2 + l*HD,
                                     vg_b1 + l*HD, vg_w2 + (size_t)l*HD, vg_b2 + l,
                                     frag + (size_t)(l*2)*8192, frag + (size_t)(l*2+1)*8192,
                                     msg, vecb, out + 67072, (l == NL-1) ? 1 : 0);
    k_su<<<NRES, HD, 0, stream>>>(msg, su_w1 + (size_t)l*2*HD*HD, su_b1 + l*HD,
                                  su_w2 + (size_t)l*HD*HD, su_b2 + l*HD,
                                  ln_g + l*HD, ln_b + l*HD, node);
  }

  k_global<<<1, HD, 0, stream>>>(node, pmk, mmk, gh_w1, gh_b1, gh_w2, gh_b2, out, gemb);
  k_cat   <<<NRES, HD, 0, stream>>>(node, vecb, gemb, pmk, mmk, cmk, ch_w1, ch_b1, ch_w2, ch_b2, out);
}
